// GCN_17197049053758
// MI455X (gfx1250) — compile-verified, matrix operand order checked by reading, not run on hardware
//
#include <hip/hip_runtime.h>

// ---------------------------------------------------------------------------
// 2-layer GCN forward for MI455X (gfx1250).
//
// Roofline: GEMMs are ~13 GFLOP (trivial); edge scatter-adds dominate.
// Reordering layer 2 as (h1*norm_src)@W2 BEFORE aggregation keeps both
// aggregations at 128-dim messages (~3.3 GB edge traffic, ~140us @ 23.3TB/s).
// GEMMs use the fast 16-bit XDL path: v_wmma_f32_16x16x32_f16 with fp32
// accumulation; activations/weights converted to f16 (values are O(1)).
// ---------------------------------------------------------------------------

typedef __attribute__((ext_vector_type(16))) _Float16 v16h;
typedef __attribute__((ext_vector_type(8)))  float    v8f;

#define DIM0 128   // message dim for both aggregations

// ---- degree accumulation -------------------------------------------------
__global__ void deg_kernel(const int* __restrict__ src, const int* __restrict__ dst,
                           float* __restrict__ degs, float* __restrict__ degd, int nE) {
  int e = blockIdx.x * blockDim.x + threadIdx.x;
  if (e >= nE) return;
  atomicAdd(&degs[src[e]], 1.0f);
  atomicAdd(&degd[dst[e]], 1.0f);
}

// ---- deg -> rsqrt(max(deg,1)) in place -----------------------------------
__global__ void norm_kernel(float* __restrict__ d, int n) {
  int i = blockIdx.x * blockDim.x + threadIdx.x;
  if (i >= n) return;
  d[i] = rsqrtf(fmaxf(d[i], 1.0f));
}

// ---- f32 row-major [K][N] -> f16 col-major [N][K] ------------------------
__global__ void wconv_kernel(const float* __restrict__ W, _Float16* __restrict__ Wt,
                             int K, int N) {
  int t = blockIdx.x * blockDim.x + threadIdx.x;
  if (t >= K * N) return;
  int k = t / N, n = t % N;
  Wt[(size_t)n * K + k] = (_Float16)W[t];
}

// ---- edge aggregation: out[dst] += feat[src] * (sscale ? sscale[src] : 1) -
// one wave per edge (32 lanes x float4 = 128 floats)
__global__ void agg_edges_kernel(const float4* __restrict__ feat4,
                                 const int* __restrict__ src,
                                 const int* __restrict__ dst,
                                 const float* __restrict__ sscale,
                                 float* __restrict__ out, long long nE) {
  long long tid = (long long)blockIdx.x * blockDim.x + threadIdx.x;
  long long e = tid >> 5;
  if (e >= nE) return;
  int c = (int)(tid & 31);
  int s = src[e], d = dst[e];
  float4 v = feat4[(size_t)s * (DIM0 / 4) + c];
  if (sscale) {
    float sc = sscale[s];
    v.x *= sc; v.y *= sc; v.z *= sc; v.w *= sc;
  }
  float* o = out + (size_t)d * DIM0 + c * 4;
  atomicAdd(o + 0, v.x);
  atomicAdd(o + 1, v.y);
  atomicAdd(o + 2, v.z);
  atomicAdd(o + 3, v.w);
}

// ---- agg1 * norm_dst -> f16 (GEMM-1 A operand), 128 cols -----------------
__global__ void scale_cvt_kernel(const float4* __restrict__ in,
                                 const float* __restrict__ rowscale,
                                 _Float16* __restrict__ outh, int nRows) {
  long long i = (long long)blockIdx.x * blockDim.x + threadIdx.x;
  long long tot = (long long)nRows * (DIM0 / 4);
  if (i >= tot) return;
  float s = rowscale[i >> 5];           // 32 float4 per row
  float4 v = in[i];
  _Float16* o = outh + i * 4;
  o[0] = (_Float16)(v.x * s);
  o[1] = (_Float16)(v.y * s);
  o[2] = (_Float16)(v.z * s);
  o[3] = (_Float16)(v.w * s);
}

// ---- out = out * norm_dst[row] + b2[col], 128 cols, in place -------------
__global__ void finalize_kernel(float4* __restrict__ out,
                                const float* __restrict__ ndst,
                                const float4* __restrict__ bias4, int nRows) {
  long long i = (long long)blockIdx.x * blockDim.x + threadIdx.x;
  long long tot = (long long)nRows * (DIM0 / 4);
  if (i >= tot) return;
  float s = ndst[i >> 5];
  float4 b = bias4[i & 31];
  float4 v = out[i];
  v.x = v.x * s + b.x;
  v.y = v.y * s + b.y;
  v.z = v.z * s + b.z;
  v.w = v.w * s + b.w;
  out[i] = v;
}

// ---- WMMA GEMM: C[M x NCOLS] = A[M x K] (f16) @ Wt (f16, col-major in LDS)
// 8 waves/block, each wave: 16-row stripe, all NCOLS/16 tiles in registers.
// EPI1: out_h = (f16)(relu(C + b1[col]) * norm_src[row])   (feeds GEMM-2)
// else: out_f = C (f32)                                     (pre-agg layer-2)
template <int K, int NCOLS, bool EPI1>
__global__ __launch_bounds__(256) void gemm_wmma_kernel(
    const _Float16* __restrict__ A, const _Float16* __restrict__ Wt,
    const float* __restrict__ bias, const float* __restrict__ rowscale,
    _Float16* __restrict__ outH, float* __restrict__ outF, int nRows) {
  __shared__ __align__(128) _Float16 sW[K * NCOLS];

  { // cooperative 64KB weight stage into LDS
    const uint4* gw = (const uint4*)Wt;
    uint4* sw = (uint4*)sW;
    const int n16 = K * NCOLS * (int)sizeof(_Float16) / 16;
    for (int i = threadIdx.x; i < n16; i += blockDim.x) sw[i] = gw[i];
  }
  __syncthreads();

  const int wave = threadIdx.x >> 5;
  const int lane = threadIdx.x & 31;
  const int rowBase = (blockIdx.x * 8 + wave) * 16;
  if (rowBase >= nRows) return;     // whole-wave tile out of range (after barrier)

  const int lr = lane & 15;         // A: row-in-tile, B: col-in-tile
  const int kh = lane >> 4;         // K-half selector (0: K 0..15, 1: K 16..31)
  constexpr int NT = NCOLS / 16;

  v8f acc[NT];
#pragma unroll
  for (int n = 0; n < NT; ++n) acc[n] = (v8f){0.f, 0.f, 0.f, 0.f, 0.f, 0.f, 0.f, 0.f};

  int arow = rowBase + lr;
  if (arow >= nRows) arow = nRows - 1;            // clamp (tiles are 16-aligned anyway)
  const _Float16* aP = A + (size_t)arow * K + kh * 16;
  const _Float16* bP = sW + (size_t)lr * K + kh * 16;

#pragma unroll
  for (int k0 = 0; k0 < K; k0 += 32) {
    v16h a = *(const v16h*)(aP + k0);
#pragma unroll
    for (int n = 0; n < NT; ++n) {
      v16h b = *(const v16h*)(bP + (size_t)n * 16 * K + k0);
      acc[n] = __builtin_amdgcn_wmma_f32_16x16x32_f16(
          false, a, false, b, (short)0, acc[n], false, false);
    }
  }

  // C layout: lane = (m>=8 ? 16 : 0) + n, vgpr j -> row (kh*8 + j), col lr
  const int rBase = rowBase + kh * 8;
  if constexpr (EPI1) {
    float ns[8];
#pragma unroll
    for (int j = 0; j < 8; ++j) {
      int r = rBase + j;
      ns[j] = (r < nRows) ? rowscale[r] : 0.f;
    }
#pragma unroll
    for (int n = 0; n < NT; ++n) {
      int col = n * 16 + lr;
      float bb = bias[col];
#pragma unroll
      for (int j = 0; j < 8; ++j) {
        int r = rBase + j;
        if (r < nRows) {
          float v = acc[n][j] + bb;
          v = v > 0.f ? v : 0.f;
          outH[(size_t)r * NCOLS + col] = (_Float16)(v * ns[j]);
        }
      }
    }
  } else {
#pragma unroll
    for (int n = 0; n < NT; ++n) {
      int col = n * 16 + lr;
#pragma unroll
      for (int j = 0; j < 8; ++j) {
        int r = rBase + j;
        if (r < nRows) outF[(size_t)r * NCOLS + col] = acc[n][j];
      }
    }
  }
}

// ---------------------------------------------------------------------------
extern "C" void kernel_launch(void* const* d_in, const int* in_sizes, int n_in,
                              void* d_out, int out_size, void* d_ws, size_t ws_size,
                              hipStream_t stream) {
  const float* x  = (const float*)d_in[0];
  const int*   src = (const int*)d_in[1];
  const int*   dst = (const int*)d_in[2];
  const float* W1 = (const float*)d_in[3];
  const float* b1 = (const float*)d_in[4];
  const float* W2 = (const float*)d_in[5];
  const float* b2 = (const float*)d_in[6];
  float* out = (float*)d_out;

  const int E = in_sizes[1];
  const int N = in_sizes[0] / 128;   // IN_DIM = 128
  const int HID = 256;

  // carve scratch (256B aligned slots)
  char* ws = (char*)d_ws;
  size_t p = 0;
  auto take = [&](size_t bytes) -> char* {
    char* r = ws + p;
    p = (p + bytes + 255) & ~(size_t)255;
    return r;
  };
  float*    nsrc = (float*)take((size_t)N * 4);              // deg_out -> norm_src
  float*    ndst = (float*)take((size_t)N * 4);              // deg_in  -> norm_dst
  float*    agg1 = (float*)take((size_t)N * 128 * 4);        // layer-1 agg, reused as g
  _Float16* a1h  = (_Float16*)take((size_t)N * 128 * 2);     // f16(agg1 * norm_dst)
  _Float16* h1s  = (_Float16*)take((size_t)N * 256 * 2);     // f16(relu(...)*norm_src)
  _Float16* W1h  = (_Float16*)take((size_t)128 * 256 * 2);   // W1^T f16
  _Float16* W2h  = (_Float16*)take((size_t)256 * 128 * 2);   // W2^T f16

  // zero accumulators (graph-capturable memset nodes)
  hipMemsetAsync(nsrc, 0, (size_t)N * 4, stream);
  hipMemsetAsync(ndst, 0, (size_t)N * 4, stream);
  hipMemsetAsync(agg1, 0, (size_t)N * 128 * 4, stream);
  hipMemsetAsync(out,  0, (size_t)N * 128 * 4, stream);

  // degrees -> norms
  deg_kernel<<<(E + 255) / 256, 256, 0, stream>>>(src, dst, nsrc, ndst, E);
  norm_kernel<<<(N + 255) / 256, 256, 0, stream>>>(nsrc, N);
  norm_kernel<<<(N + 255) / 256, 256, 0, stream>>>(ndst, N);

  // weights -> f16 col-major
  wconv_kernel<<<(128 * 256 + 255) / 256, 256, 0, stream>>>(W1, W1h, 128, HID);
  wconv_kernel<<<(256 * 128 + 255) / 256, 256, 0, stream>>>(W2, W2h, HID, 128);

  // layer-1 aggregation: agg1[d] += x[s] * norm_src[s]
  long long ethreads = (long long)E * 32;
  unsigned eblocks = (unsigned)((ethreads + 255) / 256);
  agg_edges_kernel<<<eblocks, 256, 0, stream>>>((const float4*)x, src, dst, nsrc, agg1, E);

  // a1h = f16(agg1 * norm_dst)
  long long c4 = (long long)N * 32;
  unsigned cblocks = (unsigned)((c4 + 255) / 256);
  scale_cvt_kernel<<<cblocks, 256, 0, stream>>>((const float4*)agg1, ndst, a1h, N);

  // GEMM-1: h1s = f16(relu(a1h @ W1 + b1) * norm_src)   [100k x 256]
  int gblocks = (N + 127) / 128;
  gemm_wmma_kernel<128, 256, true><<<gblocks, 256, 0, stream>>>(
      a1h, W1h, b1, nsrc, h1s, nullptr, N);

  // GEMM-2 (reordered before aggregation): g = h1s @ W2   [100k x 128] f32
  gemm_wmma_kernel<256, 128, false><<<gblocks, 256, 0, stream>>>(
      h1s, W2h, nullptr, nullptr, nullptr, agg1, N);

  // layer-2 aggregation: out[d] += g[s]
  agg_edges_kernel<<<eblocks, 256, 0, stream>>>((const float4*)agg1, src, dst, nullptr, out, E);

  // out = out * norm_dst + b2
  finalize_kernel<<<cblocks, 256, 0, stream>>>((float4*)out, ndst, (const float4*)b2, N);
}
